// CLS_30288109371814
// MI455X (gfx1250) — matrix operand order repaired and checked by reading, not compiled, not run on hardware
//
#include <hip/hip_runtime.h>
#include <hip/hip_bf16.h>
#include <math.h>

#define D_IN 2048
#define D_OUT 512

typedef __attribute__((ext_vector_type(16))) __bf16 v16bf;
typedef __attribute__((ext_vector_type(8))) float v8f;
typedef int b128_t __attribute__((vector_size(16)));  // matches builtin's V4i

#if __has_builtin(__builtin_amdgcn_global_load_async_to_lds_b128)
#define ASYNC_B 1
#endif

__device__ __forceinline__ unsigned short f2bf(float f) {
  union { float f; unsigned u; } x; x.f = f;
  unsigned r = x.u + 0x7FFFu + ((x.u >> 16) & 1u);  // round-to-nearest-even
  return (unsigned short)(r >> 16);
}

// packed f32x2 -> bf16x2 (prefers V_CVT_PK_BF16_F32)
__device__ __forceinline__ unsigned pk_bf16(float a, float b) {
#if __has_builtin(__builtin_amdgcn_cvt_pk_bf16_f32)
  typedef __attribute__((ext_vector_type(2))) __bf16 v2bf;
  union { v2bf v; unsigned u; } r;
  r.v = __builtin_amdgcn_cvt_pk_bf16_f32(a, b);
  return r.u;
#else
  return (unsigned)f2bf(a) | ((unsigned)f2bf(b) << 16);
#endif
}

__device__ __forceinline__ void wait_async0() {
#if __has_builtin(__builtin_amdgcn_s_wait_asynccnt)
  __builtin_amdgcn_s_wait_asynccnt(0);
#else
  asm volatile("s_wait_asynccnt 0x0" ::: "memory");
#endif
}

// ---- degree / normalization -------------------------------------------------
__global__ void k_deg_init(float* deg, int n) {
  int i = blockIdx.x * blockDim.x + threadIdx.x;
  if (i < n) deg[i] = 1.0f;  // self loop contributes 1
}

__global__ void k_deg_edges(const long long* ei, float* deg, int E) {
  int e = blockIdx.x * blockDim.x + threadIdx.x;
  if (e < E) atomicAdd(&deg[(int)ei[E + e]], 1.0f);
}

__global__ void k_rsqrt(float* deg, int n) {
  int i = blockIdx.x * blockDim.x + threadIdx.x;
  if (i < n) {
    float d = deg[i];
    deg[i] = d > 0.f ? rsqrtf(d) : 0.f;
  }
}

// ---- W -> bf16, transposed to column-major [n][k] ---------------------------
__global__ void k_wconv(const float* __restrict__ W, unsigned short* __restrict__ Wb) {
  int t = blockIdx.x * blockDim.x + threadIdx.x;  // D_IN*D_OUT threads
  int k = t >> 9, n = t & 511;
  Wb[(size_t)n * D_IN + k] = f2bf(W[(size_t)k * D_OUT + n]);
}

// ---- WMMA GEMM: h[N,512] = x[N,2048] @ W ------------------------------------
// BM=128, BN=256, BK=32; 8 waves as 2x4; each wave 64x64 = 4x4 WMMA tiles.
// grid.x = column blocks (2) so adjacent blocks reuse the same A rows in L2.
template <bool FULL>
__launch_bounds__(256)
__global__ void k_gemm(const float* __restrict__ x, const unsigned short* __restrict__ Wb,
                       float* __restrict__ h, int nNodes, int rowOrigin) {
  __shared__ unsigned short As[128][40];  // 128x32 bf16 A tile (+pad)
  __shared__ unsigned short Bs[256][40];  // 256 cols x 32 K bf16 B tile (+pad)

  const int t = threadIdx.x;
  const int w = t >> 5, lane = t & 31;
  const int half = lane >> 4, lid = lane & 15;
  const int mBase = (w & 1) * 64;   // wave row offset in block
  const int nBase = (w >> 1) * 64;  // wave col offset in block
  const int blockRow = rowOrigin + blockIdx.y * 128;
  const int blockCol = blockIdx.x * 256;

  v8f acc[4][4];
#pragma unroll
  for (int i = 0; i < 4; ++i)
#pragma unroll
    for (int j = 0; j < 4; ++j)
      acc[i][j] = (v8f){0.f, 0.f, 0.f, 0.f, 0.f, 0.f, 0.f, 0.f};

  for (int k0 = 0; k0 < D_IN; k0 += 32) {
    // Stage B tile first (pure bf16 copy; async -> LDS when available)
#pragma unroll
    for (int i = 0; i < 4; ++i) {
      int idx = t + i * 256;
      int nr = idx >> 2, q = idx & 3;
      const unsigned short* gp = Wb + (size_t)(blockCol + nr) * D_IN + k0 + q * 8;
#ifdef ASYNC_B
      __builtin_amdgcn_global_load_async_to_lds_b128(
          (__attribute__((address_space(1))) b128_t*)gp,
          (__attribute__((address_space(3))) b128_t*)&Bs[nr][q * 8], 0, 0);
#else
      *(uint4*)&Bs[nr][q * 8] = *(const uint4*)gp;
#endif
    }
    // Stage A tile: 128x32 f32 -> bf16 in LDS (branch-free in FULL blocks)
#pragma unroll
    for (int i = 0; i < 4; ++i) {
      int idx = t + i * 256;
      int row = idx >> 3, c4 = idx & 7;
      int grow = blockRow + row;
      float4 v = make_float4(0.f, 0.f, 0.f, 0.f);
      if (FULL || grow < nNodes)
        v = *(const float4*)(x + (size_t)grow * D_IN + k0 + c4 * 4);
      // branch-free prefetch of next K tile (wraps on last iter, clamped row)
      int pr = FULL ? grow : (grow < nNodes ? grow : nNodes - 1);
      __builtin_prefetch(x + (size_t)pr * D_IN + ((k0 + 32) & (D_IN - 1)) + c4 * 4, 0, 3);
      uint2 p;
      p.x = pk_bf16(v.x, v.y);
      p.y = pk_bf16(v.z, v.w);
      *(uint2*)&As[row][c4 * 4] = p;
    }
#ifdef ASYNC_B
    wait_async0();
#endif
    __syncthreads();

    // Build fragments per ISA 16-bit A(16x32)/B(32x16) wave32 layouts.
    v16bf afrag[4], bfrag[4];
#pragma unroll
    for (int tm = 0; tm < 4; ++tm) {
      int m = mBase + tm * 16 + lid;
      union { uint4 q[2]; v16bf v; } u;
      u.q[0] = *(const uint4*)&As[m][half * 8];       // K = half*8 + 0..7
      u.q[1] = *(const uint4*)&As[m][half * 8 + 16];  // K = 16 + half*8 + 0..7
      afrag[tm] = u.v;
    }
#pragma unroll
    for (int tn = 0; tn < 4; ++tn) {
      int n = nBase + tn * 16 + lid;
      union { uint4 q[2]; v16bf v; } u;
      u.q[0] = *(const uint4*)&Bs[n][half * 8];       // K = half*8 + 0..7
      u.q[1] = *(const uint4*)&Bs[n][half * 8 + 16];  // K = 16 + half*8 + 0..7
      bfrag[tn] = u.v;
    }

#pragma unroll
    for (int tm = 0; tm < 4; ++tm)
#pragma unroll
      for (int tn = 0; tn < 4; ++tn)
        acc[tm][tn] = __builtin_amdgcn_wmma_f32_16x16x32_bf16(
            false, afrag[tm], false, bfrag[tn], (short)0, acc[tm][tn], false, false);

    __syncthreads();
  }

  // Epilogue: C/D layout -> VGPR r holds row M = half*8 + r, col N = lid
#pragma unroll
  for (int tm = 0; tm < 4; ++tm)
#pragma unroll
    for (int tn = 0; tn < 4; ++tn)
#pragma unroll
      for (int r = 0; r < 8; ++r) {
        int gm = blockRow + mBase + tm * 16 + half * 8 + r;
        if (FULL || gm < nNodes) {
          int gc = blockCol + nBase + tn * 16 + lid;
          h[(size_t)gm * D_OUT + gc] = acc[tm][tn][r];
        }
      }
}

// ---- out = dinv^2 * h + bias (self loop + bias) -----------------------------
__global__ void k_init_out(const float* __restrict__ h, const float* __restrict__ dinv,
                           const float* __restrict__ bias, float* __restrict__ out, int n) {
  int t = blockIdx.x * blockDim.x + threadIdx.x;  // n*128 threads (float4 each)
  int i = t >> 7, q = t & 127;
  if (i >= n) return;
  float di = dinv[i];
  float s = di * di;
  float4 hv = ((const float4*)(h + (size_t)i * D_OUT))[q];
  float4 bv = ((const float4*)bias)[q];
  float4 o = make_float4(s * hv.x + bv.x, s * hv.y + bv.y, s * hv.z + bv.z, s * hv.w + bv.w);
  ((float4*)(out + (size_t)i * D_OUT))[q] = o;
}

// ---- edge scatter: one wave per edge, f32 atomics (hits L2) -----------------
__global__ void k_scatter(const long long* __restrict__ ei, const float* __restrict__ h,
                          const float* __restrict__ dinv, float* __restrict__ out, int E) {
  int gw = (blockIdx.x * blockDim.x + threadIdx.x) >> 5;
  int lane = threadIdx.x & 31;
  if (gw >= E) return;
  int s = (int)ei[gw];
  int d = (int)ei[E + gw];
  float norm = dinv[s] * dinv[d];
  const float4* hs = (const float4*)(h + (size_t)s * D_OUT);
  float* od = out + (size_t)d * D_OUT;
#pragma unroll
  for (int i = 0; i < 4; ++i) {
    int q = lane + i * 32;
    float4 v = hs[q];
    atomicAdd(od + q * 4 + 0, norm * v.x);
    atomicAdd(od + q * 4 + 1, norm * v.y);
    atomicAdd(od + q * 4 + 2, norm * v.z);
    atomicAdd(od + q * 4 + 3, norm * v.w);
  }
}

// ---- row-wise log_softmax, one wave32 per row, in place ---------------------
__global__ void k_lsm(float* __restrict__ out, int n) {
  int gw = (blockIdx.x * blockDim.x + threadIdx.x) >> 5;
  int lane = threadIdx.x & 31;
  if (gw >= n) return;
  float4* row = (float4*)(out + (size_t)gw * D_OUT);
  float4 v[4];
  float m = -INFINITY;
#pragma unroll
  for (int i = 0; i < 4; ++i) {
    v[i] = row[lane + i * 32];
    m = fmaxf(m, fmaxf(fmaxf(v[i].x, v[i].y), fmaxf(v[i].z, v[i].w)));
  }
#pragma unroll
  for (int off = 16; off > 0; off >>= 1) m = fmaxf(m, __shfl_xor(m, off, 32));
  float s = 0.f;
#pragma unroll
  for (int i = 0; i < 4; ++i)
    s += __expf(v[i].x - m) + __expf(v[i].y - m) + __expf(v[i].z - m) + __expf(v[i].w - m);
#pragma unroll
  for (int off = 16; off > 0; off >>= 1) s += __shfl_xor(s, off, 32);
  float lg = m + __logf(s);
#pragma unroll
  for (int i = 0; i < 4; ++i)
    row[lane + i * 32] = make_float4(v[i].x - lg, v[i].y - lg, v[i].z - lg, v[i].w - lg);
}

extern "C" void kernel_launch(void* const* d_in, const int* in_sizes, int n_in,
                              void* d_out, int out_size, void* d_ws, size_t ws_size,
                              hipStream_t stream) {
  const float*     x    = (const float*)d_in[0];
  const long long* ei   = (const long long*)d_in[1];   // int64 edge_index [2,E]
  const float*     W    = (const float*)d_in[2];
  const float*     bias = (const float*)d_in[3];
  float*           out  = (float*)d_out;

  const int N = in_sizes[0] / D_IN;  // 50000
  const int E = in_sizes[1] / 2;     // 400000

  // workspace layout: h [N*512 f32] | dinv [N f32] | Wb [512*2048 bf16]
  float* h = (float*)d_ws;
  float* dinv = h + (size_t)N * D_OUT;
  unsigned short* Wb = (unsigned short*)(dinv + N);

  k_deg_init<<<(N + 255) / 256, 256, 0, stream>>>(dinv, N);
  k_deg_edges<<<(E + 255) / 256, 256, 0, stream>>>(ei, dinv, E);
  k_rsqrt<<<(N + 255) / 256, 256, 0, stream>>>(dinv, N);
  k_wconv<<<(D_IN * D_OUT) / 256, 256, 0, stream>>>(W, Wb);

  const int fullRows = N / 128;          // bound-check-free row blocks
  const int rem = N - fullRows * 128;    // tail rows
  if (fullRows > 0) {
    dim3 gf(D_OUT / 256, fullRows);
    k_gemm<true><<<gf, 256, 0, stream>>>(x, Wb, h, N, 0);
  }
  if (rem > 0) {
    dim3 ge(D_OUT / 256, 1);
    k_gemm<false><<<ge, 256, 0, stream>>>(x, Wb, h, N, fullRows * 128);
  }

  k_init_out<<<(N * 128 + 255) / 256, 256, 0, stream>>>(h, dinv, bias, out, N);
  k_scatter<<<(E * 32 + 255) / 256, 256, 0, stream>>>(ei, h, dinv, out, E);
  k_lsm<<<(N * 32 + 255) / 256, 256, 0, stream>>>(out, N);
}
